// SelfAttention_15857019257221
// MI455X (gfx1250) — matrix operand order repaired and checked by reading, not compiled, not run on hardware
//
#include <hip/hip_runtime.h>
#include <hip/hip_bf16.h>

// Sizes for this problem
#define BATCH  4
#define SEQ    4096
#define EMBED  1024
#define HD     128
#define QKVN   384          // 3 * HD
#define ROWS   (BATCH*SEQ)  // 16384

typedef __attribute__((ext_vector_type(16))) __bf16 v16bf;
typedef __attribute__((ext_vector_type(8)))  float  v8f;
typedef __attribute__((ext_vector_type(4)))  int    v4i;

union Frag { v16bf v; uint4 u[2]; };

// ---- bf16 conversion: native cast (one v_cvt) instead of manual RNE ----
__device__ __forceinline__ unsigned short f2bf(float f) {
  __bf16 h = (__bf16)f;
  return __builtin_bit_cast(unsigned short, h);
}

// packed pair {lo, hi} -> one u32 (v_cvt_pk_bf16_f32 when available)
__device__ __forceinline__ unsigned int pk2bf(float lo, float hi) {
#if __has_builtin(__builtin_amdgcn_cvt_pk_bf16_f32)
  auto t = __builtin_amdgcn_cvt_pk_bf16_f32(lo, hi);
  return __builtin_bit_cast(unsigned int, t);
#else
  return (unsigned int)f2bf(lo) | ((unsigned int)f2bf(hi) << 16);
#endif
}

// ---- async global -> LDS copy (ASYNCcnt path) with safe fallback ----
#if __has_builtin(__builtin_amdgcn_global_load_async_to_lds_b128)
#define ASYNC_LDS 1
#else
#define ASYNC_LDS 0
#endif

__device__ __forceinline__ void cp16_g2l(unsigned short* dst, const unsigned short* src) {
#if ASYNC_LDS
  __builtin_amdgcn_global_load_async_to_lds_b128(
      (__attribute__((address_space(1))) v4i*)src,
      (__attribute__((address_space(3))) v4i*)dst, 0, 0);
#else
  *(uint4*)dst = *(const uint4*)src;
#endif
}

__device__ __forceinline__ void wait_async() {
#if ASYNC_LDS
#if __has_builtin(__builtin_amdgcn_s_wait_asynccnt)
  __builtin_amdgcn_s_wait_asynccnt(0);
#else
  asm volatile("s_wait_asynccnt 0x0" ::: "memory");
#endif
#endif
}

#define WMMA_BF16(A, B, C) \
  __builtin_amdgcn_wmma_f32_16x16x32_bf16(false, (A), false, (B), (short)0, (C), false, false)

// ---------------------------------------------------------------------------
// Kernel 1: qkv[16384][384] (bf16) = x[16384][1024] (f32) @ w_qkv[1024][384]
// ---------------------------------------------------------------------------
__global__ __launch_bounds__(256) void qkv_gemm(const float* __restrict__ x,
                                                const float* __restrict__ w,
                                                unsigned short* __restrict__ qkv) {
  __shared__ __align__(16) unsigned short As[128][40];  // [m][k], pad 8
  __shared__ __align__(16) unsigned short Bs[128][40];  // [n][k] transposed, pad 8

  const int tid  = threadIdx.x;
  const int lane = tid & 31, wave = tid >> 5;
  const int half = lane >> 4, l16 = lane & 15;
  const int wm = wave >> 2, wn = wave & 3;              // 2 x 4 wave grid
  const int m0 = blockIdx.y * 128;
  const int n0 = blockIdx.x * 128;

  v8f acc[4][2];
  for (int i = 0; i < 4; i++)
    for (int j = 0; j < 2; j++)
      acc[i][j] = (v8f){0.f,0.f,0.f,0.f,0.f,0.f,0.f,0.f};

  const int arow = tid >> 1, acol = (tid & 1) * 16;     // A stage: [128][32]
  const int bkp  = tid >> 4;                            // B stage: k-pair 0..15
  const int bng  = (tid & 15) * 8;                      // B stage: n offset 0..120

  for (int kt = 0; kt < EMBED / 32; ++kt) {
    // --- stage A tile (f32 -> bf16, packed pairs, 2 x ds_store_b128) ---
    {
      const float* src = x + (size_t)(m0 + arow) * EMBED + kt * 32 + acol;
      float4 f0 = *(const float4*)(src);
      float4 f1 = *(const float4*)(src + 4);
      float4 f2 = *(const float4*)(src + 8);
      float4 f3 = *(const float4*)(src + 12);
      uint4 pa, pb;
      pa.x = pk2bf(f0.x, f0.y); pa.y = pk2bf(f0.z, f0.w);
      pa.z = pk2bf(f1.x, f1.y); pa.w = pk2bf(f1.z, f1.w);
      pb.x = pk2bf(f2.x, f2.y); pb.y = pk2bf(f2.z, f2.w);
      pb.z = pk2bf(f3.x, f3.y); pb.w = pk2bf(f3.z, f3.w);
      *(uint4*)&As[arow][acol]     = pa;
      *(uint4*)&As[arow][acol + 8] = pb;
      if (kt + 1 < EMBED / 32)
        __builtin_prefetch(src + 32, 0, 1);
    }
    // --- stage B tile transposed (k-pairs packed into b32 LDS stores) ---
    {
      const float* s0 = w + (size_t)(kt * 32 + 2 * bkp) * QKVN + n0 + bng;
      const float* s1 = s0 + QKVN;
      float4 a0 = *(const float4*)s0, a1 = *(const float4*)(s0 + 4);
      float4 b0 = *(const float4*)s1, b1 = *(const float4*)(s1 + 4);
      float r0[8] = {a0.x,a0.y,a0.z,a0.w, a1.x,a1.y,a1.z,a1.w};
      float r1[8] = {b0.x,b0.y,b0.z,b0.w, b1.x,b1.y,b1.z,b1.w};
      for (int e = 0; e < 8; e++)
        *(unsigned int*)&Bs[bng + e][2 * bkp] = pk2bf(r0[e], r1[e]);
    }
    __syncthreads();

    // --- compute: 4 M-frags x 2 N-frags per wave ---
    Frag a[4];
    for (int i = 0; i < 4; i++) {
      int r = wm * 64 + i * 16 + l16;
      a[i].u[0] = *(const uint4*)&As[r][half * 8];
      a[i].u[1] = *(const uint4*)&As[r][16 + half * 8];
    }
    for (int j = 0; j < 2; j++) {
      Frag b;
      int n = wn * 32 + j * 16 + l16;
      b.u[0] = *(const uint4*)&Bs[n][half * 8];
      b.u[1] = *(const uint4*)&Bs[n][16 + half * 8];
      for (int i = 0; i < 4; i++)
        acc[i][j] = WMMA_BF16(a[i].v, b.v, acc[i][j]);
    }
    __syncthreads();
  }

  // --- store bf16 ---
  for (int i = 0; i < 4; i++)
    for (int j = 0; j < 2; j++) {
      int col  = n0 + wn * 32 + j * 16 + l16;
      int rowb = m0 + wm * 64 + i * 16 + half * 8;
      for (int r = 0; r < 8; r++)
        qkv[(size_t)(rowb + r) * QKVN + col] = f2bf(acc[i][j][r]);
    }
}

// ---------------------------------------------------------------------------
// Kernel 2: causal flash attention over bf16 q/k/v packed in qkv[.][384]
// out: attout[16384][128] bf16
// ---------------------------------------------------------------------------
__global__ __launch_bounds__(256) void attention(const unsigned short* __restrict__ qkv,
                                                 unsigned short* __restrict__ attout) {
  __shared__ __align__(16) unsigned short Ks[32][136];  // [key][d], pad 8
  __shared__ __align__(16) unsigned short Vs[128][40];  // [d][key] transposed, pad 8
  __shared__ __align__(16) unsigned short Ps[8][16][32];// per-wave P tile

  const int tid  = threadIdx.x;
  const int lane = tid & 31, wave = tid >> 5;
  const int half = lane >> 4, l16 = lane & 15;
  const int b  = blockIdx.y;
  const int qt = blockIdx.x;
  const size_t base = (size_t)b * SEQ * QKVN;
  const int qrow0 = qt * 128 + wave * 16;               // this wave's 16 query rows

  // --- Q fragments held in registers: 4 x (16x32) over head dim 128 ---
  Frag qa[4];
  {
    const unsigned short* qrow = qkv + base + (size_t)(qrow0 + l16) * QKVN;
    for (int j = 0; j < 4; j++) {
      qa[j].u[0] = *(const uint4*)(qrow + j * 32 + half * 8);
      qa[j].u[1] = *(const uint4*)(qrow + j * 32 + 16 + half * 8);
    }
  }

  v8f o[8];
  for (int f = 0; f < 8; f++) o[f] = (v8f){0.f,0.f,0.f,0.f,0.f,0.f,0.f,0.f};
  float mrow[8], lrow[8];
  for (int r = 0; r < 8; r++) { mrow[r] = -INFINITY; lrow[r] = 0.f; }

  const float scale = 0.03125f;        // 1024^-0.5 (reference scales by embed_dim)
  const float LOG2E = 1.44269504f;
  const int srow = tid >> 3, scol = (tid & 7) * 16;     // K stage coords [32][128]
  const int vkp  = tid >> 4;                            // V stage: key pair 0..15
  const int vdg  = (tid & 15) * 8;                      // V stage: d offset 0..120
  const int nkt  = 4 * (qt + 1);                        // causal: keys <= query tile end

  for (int kt = 0; kt < nkt; ++kt) {
    // --- stage K tile [32 keys][128] via async global->LDS copies ---
    {
      const unsigned short* src = qkv + base + (size_t)(kt * 32 + srow) * QKVN + HD + scol;
      cp16_g2l(&Ks[srow][scol],     src);
      cp16_g2l(&Ks[srow][scol + 8], src + 8);
    }
    // --- stage V tile transposed: Vs[d][key], packed key-pairs ---
    {
      const unsigned short* s0 = qkv + base + (size_t)(kt * 32 + 2 * vkp) * QKVN + 2 * HD + vdg;
      const unsigned short* s1 = s0 + QKVN;
      uint4 u0 = *(const uint4*)s0;     // 8 bf16 of key 2*vkp, d = vdg..vdg+7
      uint4 u1 = *(const uint4*)s1;     // 8 bf16 of key 2*vkp+1
      unsigned int ka[4] = {u0.x, u0.y, u0.z, u0.w};
      unsigned int kb[4] = {u1.x, u1.y, u1.z, u1.w};
      for (int j = 0; j < 4; j++) {
        unsigned int lo = (ka[j] & 0xFFFFu) | (kb[j] << 16);
        unsigned int hi = (ka[j] >> 16) | (kb[j] & 0xFFFF0000u);
        *(unsigned int*)&Vs[vdg + 2 * j][2 * vkp]     = lo;
        *(unsigned int*)&Vs[vdg + 2 * j + 1][2 * vkp] = hi;
      }
    }
    wait_async();
    __syncthreads();

    // --- S = Q K^T : 2 key-column fragments, 4-step K-dim chain ---
    v8f s[2];
    s[0] = (v8f){0.f,0.f,0.f,0.f,0.f,0.f,0.f,0.f};
    s[1] = (v8f){0.f,0.f,0.f,0.f,0.f,0.f,0.f,0.f};
    for (int n = 0; n < 2; n++) {
      for (int j = 0; j < 4; j++) {
        Frag kbf;
        int krow = n * 16 + l16;
        int dc   = j * 32 + half * 8;
        kbf.u[0] = *(const uint4*)&Ks[krow][dc];
        kbf.u[1] = *(const uint4*)&Ks[krow][dc + 16];
        s[n] = WMMA_BF16(qa[j].v, kbf.v, s[n]);
      }
    }

    // --- causal mask + scale + online softmax ---
    float p[2][8];
    for (int n = 0; n < 2; n++) {
      int kg = kt * 32 + n * 16 + l16;
      for (int r = 0; r < 8; r++) {
        int qg  = qrow0 + 8 * half + r;
        float v = s[n][r] * scale;
        if (kg > qg) v = -INFINITY;
        p[n][r] = v;
      }
    }
    for (int r = 0; r < 8; r++) {
      float t = fmaxf(p[0][r], p[1][r]);
      for (int msk = 8; msk >= 1; msk >>= 1) t = fmaxf(t, __shfl_xor(t, msk, 32));
      float mn = fmaxf(mrow[r], t);
      float alpha = exp2f((mrow[r] - mn) * LOG2E);
      mrow[r] = mn;
      float e0 = exp2f((p[0][r] - mn) * LOG2E);
      float e1 = exp2f((p[1][r] - mn) * LOG2E);
      float sum = e0 + e1;
      for (int msk = 8; msk >= 1; msk >>= 1) sum += __shfl_xor(sum, msk, 32);
      lrow[r] = lrow[r] * alpha + sum;
      for (int f = 0; f < 8; f++) o[f][r] *= alpha;
      // C-layout -> LDS (per-wave region; LDS ops are in-order per wave)
      Ps[wave][8 * half + r][l16]      = f2bf(e0);
      Ps[wave][8 * half + r][16 + l16] = f2bf(e1);
    }

    // --- O += P @ V ---
    Frag pa;
    pa.u[0] = *(const uint4*)&Ps[wave][l16][half * 8];
    pa.u[1] = *(const uint4*)&Ps[wave][l16][16 + half * 8];
    for (int f = 0; f < 8; f++) {
      Frag vb;
      int vr = f * 16 + l16;
      vb.u[0] = *(const uint4*)&Vs[vr][half * 8];
      vb.u[1] = *(const uint4*)&Vs[vr][16 + half * 8];
      o[f] = WMMA_BF16(pa.v, vb.v, o[f]);
    }
    __syncthreads();
  }

  // --- epilogue: O / l, store bf16 ---
  for (int f = 0; f < 8; f++) {
    for (int r = 0; r < 8; r++) {
      int qg = qrow0 + 8 * half + r;
      attout[((size_t)b * SEQ + qg) * HD + f * 16 + l16] =
          f2bf(o[f][r] * (1.f / lrow[r]));
    }
  }
}

// ---------------------------------------------------------------------------
// Kernel 3: y[16384][1024] (f32) = attout[16384][128] (bf16) @ w_out + b_out
// ---------------------------------------------------------------------------
__global__ __launch_bounds__(256) void out_gemm(const unsigned short* __restrict__ a,
                                                const float* __restrict__ w,
                                                const float* __restrict__ bias,
                                                float* __restrict__ y) {
  __shared__ __align__(16) unsigned short As[128][40];
  __shared__ __align__(16) unsigned short Bs[128][40];

  const int tid  = threadIdx.x;
  const int lane = tid & 31, wave = tid >> 5;
  const int half = lane >> 4, l16 = lane & 15;
  const int wm = wave >> 2, wn = wave & 3;
  const int m0 = blockIdx.y * 128;
  const int n0 = blockIdx.x * 128;

  v8f acc[4][2];
  for (int i = 0; i < 4; i++)
    for (int j = 0; j < 2; j++)
      acc[i][j] = (v8f){0.f,0.f,0.f,0.f,0.f,0.f,0.f,0.f};

  const int arow = tid >> 1, acol = (tid & 1) * 16;
  const int bkp  = tid >> 4;
  const int bng  = (tid & 15) * 8;

  for (int kt = 0; kt < HD / 32; ++kt) {
    // --- stage A tile (already bf16) via async global->LDS copies ---
    {
      const unsigned short* src = a + (size_t)(m0 + arow) * HD + kt * 32 + acol;
      cp16_g2l(&As[arow][acol],     src);
      cp16_g2l(&As[arow][acol + 8], src + 8);
    }
    // --- stage B tile transposed (f32 -> bf16, packed k-pairs) ---
    {
      const float* s0 = w + (size_t)(kt * 32 + 2 * bkp) * EMBED + n0 + bng;
      const float* s1 = s0 + EMBED;
      float4 a0 = *(const float4*)s0, a1 = *(const float4*)(s0 + 4);
      float4 b0 = *(const float4*)s1, b1 = *(const float4*)(s1 + 4);
      float r0[8] = {a0.x,a0.y,a0.z,a0.w, a1.x,a1.y,a1.z,a1.w};
      float r1[8] = {b0.x,b0.y,b0.z,b0.w, b1.x,b1.y,b1.z,b1.w};
      for (int e = 0; e < 8; e++)
        *(unsigned int*)&Bs[bng + e][2 * bkp] = pk2bf(r0[e], r1[e]);
    }
    wait_async();
    __syncthreads();

    Frag af[4];
    for (int i = 0; i < 4; i++) {
      int r = wm * 64 + i * 16 + l16;
      af[i].u[0] = *(const uint4*)&As[r][half * 8];
      af[i].u[1] = *(const uint4*)&As[r][16 + half * 8];
    }
    for (int j = 0; j < 2; j++) {
      Frag b;
      int n = wn * 32 + j * 16 + l16;
      b.u[0] = *(const uint4*)&Bs[n][half * 8];
      b.u[1] = *(const uint4*)&Bs[n][16 + half * 8];
      for (int i = 0; i < 4; i++)
        acc[i][j] = WMMA_BF16(af[i].v, b.v, acc[i][j]);
    }
    __syncthreads();
  }

  for (int j = 0; j < 2; j++) {
    int col = n0 + wn * 32 + j * 16 + l16;
    float bv = bias[col];
    for (int i = 0; i < 4; i++) {
      int rowb = m0 + wm * 64 + i * 16 + half * 8;
      for (int r = 0; r < 8; r++)
        y[(size_t)(rowb + r) * EMBED + col] = acc[i][j][r] + bv;
    }
  }
}

// ---------------------------------------------------------------------------
extern "C" void kernel_launch(void* const* d_in, const int* in_sizes, int n_in,
                              void* d_out, int out_size, void* d_ws, size_t ws_size,
                              hipStream_t stream) {
  const float* x     = (const float*)d_in[0];
  const float* w_qkv = (const float*)d_in[1];
  const float* w_out = (const float*)d_in[2];
  const float* b_out = (const float*)d_in[3];
  float* out = (float*)d_out;

  unsigned short* qkv    = (unsigned short*)d_ws;                 // 16384*384 bf16
  unsigned short* attout = qkv + (size_t)ROWS * QKVN;             // 16384*128 bf16

  // 1) qkv = x @ w_qkv (bf16 out)
  qkv_gemm<<<dim3(QKVN / 128, ROWS / 128), 256, 0, stream>>>(x, w_qkv, qkv);
  // 2) causal flash attention
  attention<<<dim3(SEQ / 128, BATCH), 256, 0, stream>>>(qkv, attout);
  // 3) y = attout @ w_out + b_out (f32 out)
  out_gemm<<<dim3(EMBED / 128, ROWS / 128), 256, 0, stream>>>(attout, w_out, b_out, out);
}
